// PathAgg_att_sample_layer_14774687498275
// MI455X (gfx1250) — matrix operand order repaired and checked by reading, not compiled, not run on hardware
//
#include <hip/hip_runtime.h>
#include <hip/hip_bf16.h>
#include <math.h>

#define N_NODES 20000
#define P_PATHS 400000
#define L_LEN   5
#define D_DIM   64
#define G_DIM   192   // 3*D
#define H_HEADS 4

typedef __attribute__((ext_vector_type(16))) __bf16          v16bf;
typedef __attribute__((ext_vector_type(8)))  float           v8f;

union BFrag { v16bf b; uint4 q[2]; };

__device__ __forceinline__ v8f wmma_bf16(v16bf a, v16bf b, v8f c) {
  // D = A(16x32 bf16) * B(32x16 bf16) + C(16x16 f32)
  return __builtin_amdgcn_wmma_f32_16x16x32_bf16(
      /*neg_a=*/false, a, /*neg_b=*/false, b,
      /*c_mod=*/(short)0, c, /*reuse_a=*/false, /*reuse_b=*/false);
}

// v_exp_f32 + v_rcp_f32 paths (no div_scale/Newton chains, no OCML tanh)
__device__ __forceinline__ float sigmoid_fast(float v) {
  return __builtin_amdgcn_rcpf(1.f + __expf(-v));
}
__device__ __forceinline__ float tanh_fast(float v) {
  // 1 - 2/(e^{2v}+1): saturates correctly (+1 / -1) at the exp limits
  return 1.f - 2.f * __builtin_amdgcn_rcpf(__expf(2.f * v) + 1.f);
}

__global__ void pathagg_init(float* __restrict__ out, float* __restrict__ denom) {
  int i = blockIdx.x * blockDim.x + threadIdx.x;
  int stride = gridDim.x * blockDim.x;
  for (int k = i; k < N_NODES * 256; k += stride) out[k] = 0.f;
  for (int k = i; k < N_NODES * H_HEADS; k += stride) denom[k] = 0.f;
}

__global__ __launch_bounds__(128) void pathagg_gru_wmma(
    const float* __restrict__ x,   const float* __restrict__ Wih,
    const float* __restrict__ Whh, const float* __restrict__ bih,
    const float* __restrict__ bhh, const float* __restrict__ a,
    const int*   __restrict__ pl,  float* __restrict__ out,
    float* __restrict__ denom)
{
  __shared__ __bf16 sWih[G_DIM * D_DIM];   // bf16 weights, row-major [g][k]
  __shared__ __bf16 sWhh[G_DIM * D_DIM];
  __shared__ float sBih[G_DIM], sBhh[G_DIM];
  __shared__ float sA[D_DIM * H_HEADS];
  __shared__ float sH[4][16 * D_DIM];      // per-wave h staging (f32)
  __shared__ float sAtt[4][16 * H_HEADS];
  __shared__ int   sSeg[4][16];

  const int tid = threadIdx.x;
  for (int i = tid; i < G_DIM * D_DIM; i += 128) {
    sWih[i] = (__bf16)Wih[i];
    sWhh[i] = (__bf16)Whh[i];
  }
  for (int i = tid; i < G_DIM; i += 128) { sBih[i] = bih[i]; sBhh[i] = bhh[i]; }
  for (int i = tid; i < D_DIM * H_HEADS; i += 128) sA[i] = a[i];
  __syncthreads();

  const int wave = tid >> 5;
  const int lane = tid & 31;
  const int m16  = lane & 15;
  const int hi   = lane >> 4;      // lane half (K split per ISA A/B layouts)
  const int hi8  = hi * 8;
  const int ptile = blockIdx.x * 64 + wave * 16;   // 16 paths per wave

  float* hst = sH[wave];
  for (int i = lane; i < 16 * D_DIM; i += 32) hst[i] = 0.f;  // h0 = 0

  // A-fragment (16x32 bf16, ISA 7.12.2): lane<16 holds K {kb..kb+7, kb+16..kb+23}
  // of row m16; lane>=16 holds K {kb+8..kb+15, kb+24..kb+31}.
  auto loadA = [&](const float* row, int kb) -> v16bf {
    const float4 f0 = *(const float4*)(row + kb + hi8);
    const float4 f1 = *(const float4*)(row + kb + hi8 + 4);
    const float4 f2 = *(const float4*)(row + kb + 16 + hi8);
    const float4 f3 = *(const float4*)(row + kb + 16 + hi8 + 4);
    v16bf b;
    b[0]=(__bf16)f0.x; b[1]=(__bf16)f0.y; b[2]=(__bf16)f0.z; b[3]=(__bf16)f0.w;
    b[4]=(__bf16)f1.x; b[5]=(__bf16)f1.y; b[6]=(__bf16)f1.z; b[7]=(__bf16)f1.w;
    b[8]=(__bf16)f2.x; b[9]=(__bf16)f2.y; b[10]=(__bf16)f2.z; b[11]=(__bf16)f2.w;
    b[12]=(__bf16)f3.x; b[13]=(__bf16)f3.y; b[14]=(__bf16)f3.z; b[15]=(__bf16)f3.w;
    return b;
  };

  // B-fragment (32x16 bf16): column n = m16 <- W row (gbase+n); lane halves split K.
  // W row-major [g][k] -> each lane reads K {kb+hi8..+7, kb+16+hi8..+7} (2x16B) from LDS.
  auto loadB = [&](const __bf16* W, int gbase, int kb) -> v16bf {
    const __bf16* p = W + (gbase + m16) * D_DIM + kb + hi8;
    BFrag f;
    f.q[0] = *(const uint4*)(p);
    f.q[1] = *(const uint4*)(p + 16);
    return f.b;
  };

  #pragma unroll 1
  for (int l = 0; l < L_LEN; ++l) {
    const int node = pl[(ptile + m16) * L_LEN + l];
    const float* xr = x + (size_t)node * D_DIM;
    const v16bf ax0 = loadA(xr, 0);
    const v16bf ax1 = loadA(xr, 32);
    const float* hrow = hst + m16 * D_DIM;
    const v16bf ah0 = loadA(hrow, 0);
    const v16bf ah1 = loadA(hrow, 32);

    // Rolled over column tiles to keep VGPR pressure < 256 (occupancy!).
    // Each iteration touches only columns [t*16, t*16+16) of hst, and all
    // A-fragments were read above, so the in-iteration hst update is safe
    // (DS ops are in-order within a wave).
    #pragma unroll 1
    for (int t = 0; t < 4; ++t) {
      const int gR = t * 16 + m16;
      const int gZ = (t + 4) * 16 + m16;
      const int gN = (t + 8) * 16 + m16;
      const float bR  = sBih[gR] + sBhh[gR];
      const float bZ  = sBih[gZ] + sBhh[gZ];
      const float bXn = sBih[gN];
      const float bHn = sBhh[gN];
      v8f accR, accZ, accXn, accHn;
      #pragma unroll
      for (int e = 0; e < 8; ++e) { accR[e]=bR; accZ[e]=bZ; accXn[e]=bXn; accHn[e]=bHn; }

      // r-gate: x@Wih + h@Whh over K=64 (2 chunks of 32)
      accR  = wmma_bf16(ax0, loadB(sWih, t * 16, 0),  accR);
      accR  = wmma_bf16(ax1, loadB(sWih, t * 16, 32), accR);
      accR  = wmma_bf16(ah0, loadB(sWhh, t * 16, 0),  accR);
      accR  = wmma_bf16(ah1, loadB(sWhh, t * 16, 32), accR);
      // z-gate
      accZ  = wmma_bf16(ax0, loadB(sWih, (t + 4) * 16, 0),  accZ);
      accZ  = wmma_bf16(ax1, loadB(sWih, (t + 4) * 16, 32), accZ);
      accZ  = wmma_bf16(ah0, loadB(sWhh, (t + 4) * 16, 0),  accZ);
      accZ  = wmma_bf16(ah1, loadB(sWhh, (t + 4) * 16, 32), accZ);
      // n-gate: xn and hn kept separate (reference applies r only to hn)
      accXn = wmma_bf16(ax0, loadB(sWih, (t + 8) * 16, 0),  accXn);
      accXn = wmma_bf16(ax1, loadB(sWih, (t + 8) * 16, 32), accXn);
      accHn = wmma_bf16(ah0, loadB(sWhh, (t + 8) * 16, 0),  accHn);
      accHn = wmma_bf16(ah1, loadB(sWhh, (t + 8) * 16, 32), accHn);

      #pragma unroll
      for (int e = 0; e < 8; ++e) {
        const int hidx = (e + hi8) * D_DIM + t * 16 + m16;
        const float hOld = hst[hidx];
        const float r  = sigmoid_fast(accR[e]);
        const float z  = sigmoid_fast(accZ[e]);
        const float ng = tanh_fast(accXn[e] + r * accHn[e]);
        hst[hidx] = (1.f - z) * ng + z * hOld;
      }
    }
  }

  // ---- attention epilogue (denominator factored out of the segment sum) ----
  if (lane < 16)
    sSeg[wave][lane] = pl[(ptile + lane) * L_LEN + (L_LEN - 1)];

  for (int idx = lane; idx < 16 * H_HEADS; idx += 32) {
    const int m = idx >> 2, hd = idx & 3;
    const float* hr2 = hst + m * D_DIM;
    float dot = 0.f;
    #pragma unroll
    for (int d2 = 0; d2 < D_DIM; ++d2) dot += hr2[d2] * sA[d2 * H_HEADS + hd];
    const float au = __expf(dot);
    sAtt[wave][idx] = au;
    atomicAdd(&denom[(size_t)sSeg[wave][m] * H_HEADS + hd], au);
  }

  for (int idx = lane; idx < 16 * 256; idx += 32) {
    const int m = idx >> 8, r = idx & 255;
    const int hd = r >> 6, d2 = r & 63;
    const float v = sAtt[wave][m * H_HEADS + hd] * hst[m * D_DIM + d2];
    atomicAdd(&out[(size_t)sSeg[wave][m] * 256 + r], v);
  }
}

__global__ void pathagg_norm(float* __restrict__ out, const float* __restrict__ denom) {
  int i = blockIdx.x * blockDim.x + threadIdx.x;
  if (i < N_NODES * 256) {
    float dn = denom[(i >> 8) * H_HEADS + ((i >> 6) & 3)];
    out[i] = (dn != 0.f) ? out[i] / dn : 0.f;
  }
}

extern "C" void kernel_launch(void* const* d_in, const int* in_sizes, int n_in,
                              void* d_out, int out_size, void* d_ws, size_t ws_size,
                              hipStream_t stream) {
  const float* x   = (const float*)d_in[0];
  const float* Wih = (const float*)d_in[1];
  const float* Whh = (const float*)d_in[2];
  const float* bih = (const float*)d_in[3];
  const float* bhh = (const float*)d_in[4];
  const float* a   = (const float*)d_in[5];
  const int*   pl  = (const int*)d_in[6];
  float* out   = (float*)d_out;
  float* denom = (float*)d_ws;            // N*4 f32 = 320 KB

  hipLaunchKernelGGL(pathagg_init, dim3(512), dim3(256), 0, stream, out, denom);
  hipLaunchKernelGGL(pathagg_gru_wmma, dim3(P_PATHS / 64), dim3(128), 0, stream,
                     x, Wih, Whh, bih, bhh, a, pl, out, denom);
  hipLaunchKernelGGL(pathagg_norm, dim3((N_NODES * 256 + 255) / 256), dim3(256), 0,
                     stream, out, denom);
}
